// SelfAttention_Conv2D_73907797230223
// MI455X (gfx1250) — matrix operand order repaired and checked by reading, not compiled, not run on hardware
//
#include <hip/hip_runtime.h>
#include <hip/hip_bf16.h>

// ---------------------------------------------------------------------------
// Self-attention over 64x64x128 feature maps, B=4.  N = 4096, C = 128.
//   Kernel 0: W* -> f16 transposed weights WT[n][k]  (one-time, 128 KB)
//   Kernel 1: f,g = x@W (row-major f16), v = x@Wh stored TRANSPOSED vT[c][n]
//             (C-layout lanes hold 8 consecutive M at fixed N -> 16B stores)
//   Kernel 2: per (batch, 16-query-row) block:
//       S = f g^T   -> 16x4096 f16 panel entirely in LDS (128 KB)
//       P = exp(S-rowmax) in place, rowsum kept (1/sum folded into phase 4)
//       ctx = P @ v  (B-fragments contiguous from vT, ds_add_f32 accumulate)
//       o = (gamma/rowsum)*ctx @ Wv + bv + x  (WMMA + residual, fp32 out)
// All WMMA operand fetches on the hot path are contiguous 16-byte f16 loads.
// ---------------------------------------------------------------------------

#define BATCH 4
#define NPIX  4096      // 64*64
#define CH    128

typedef __attribute__((ext_vector_type(16))) _Float16 v16h;
typedef __attribute__((ext_vector_type(8)))  float    v8f;

__device__ __forceinline__ v8f wmma16(v16h a, v16h b, v8f c) {
    // (neg_a, A, neg_b, B, c_mod, C, reuse_a, reuse_b)
    return __builtin_amdgcn_wmma_f32_16x16x32_f16(false, a, false, b, (short)0, c,
                                                  false, false);
}

// 16 contiguous f16 -> B fragment (lane-local; compiler fuses to 2x b128).
__device__ __forceinline__ v16h load16h(const _Float16* __restrict__ p) {
    v16h b;
#pragma unroll
    for (int e = 0; e < 16; ++e) b[e] = p[e];
    return b;
}

// A-matrix 16x32 f16 fragment from f16 row-major source (global or LDS).
// lanes 0-15/16-31 hold rows M = lane&15; half = lane>>4:
// elems 0..7 = K[k0+half*8+e], elems 8..15 = K[k0+16+half*8+e].
__device__ __forceinline__ v16h load_A_f16(const _Float16* __restrict__ base,
                                           int ld, int row0, int k0, int lane) {
    const int r = lane & 15, half = lane >> 4;
    const _Float16* p = base + (size_t)(row0 + r) * ld;
    const int klo = k0 + half * 8;
    const int khi = k0 + 16 + half * 8;
    v16h a;
#pragma unroll
    for (int e = 0; e < 8; ++e) {
        a[e]     = p[klo + e];
        a[8 + e] = p[khi + e];
    }
    return a;
}

// A fragment from fp32 source (x) with f16 convert.
__device__ __forceinline__ v16h load_A_f32(const float* __restrict__ base, int ld,
                                           int row0, int k0, int lane) {
    const int r = lane & 15, half = lane >> 4;
    const float* p = base + (size_t)(row0 + r) * ld;
    const int klo = k0 + half * 8;
    const int khi = k0 + 16 + half * 8;
    v16h a;
#pragma unroll
    for (int e = 0; e < 8; ++e) {
        a[e]     = (_Float16)p[klo + e];
        a[8 + e] = (_Float16)p[khi + e];
    }
    return a;
}

// B fragment from a TRANSPOSED f16 matrix T[n][k] (ld = k-stride):
// col = lane&15 fixed, elems 0..7 = K[k0+half*8+e], elems 8..15 = K[k0+16+half*8+e].
__device__ __forceinline__ v16h load_B_T(const _Float16* __restrict__ T, int ld,
                                         int k0, int n0, int lane) {
    const int col = n0 + (lane & 15);
    const _Float16* p = T + (size_t)col * ld + k0 + ((lane >> 4) << 3);
    v16h b;
#pragma unroll
    for (int e = 0; e < 8; ++e) {
        b[e]     = p[e];
        b[8 + e] = p[16 + e];
    }
    return b;
}

// ---------------------------------------------------------------------------
// Kernel 0: transpose + f16-convert the four 128x128 weight matrices.
// WT layout: [mat][n][k], mat order f,g,h,v.
// ---------------------------------------------------------------------------
__global__ __launch_bounds__(256) void transpose_w_kernel(
    const float* __restrict__ Wf, const float* __restrict__ Wg,
    const float* __restrict__ Wh, const float* __restrict__ Wv,
    _Float16* __restrict__ WT) {
    const int idx = blockIdx.x * 256 + threadIdx.x;   // 4*128*128 = 65536
    const int mat = idx >> 14;
    const int n   = (idx >> 7) & 127;
    const int k   = idx & 127;
    const float* W = (mat == 0) ? Wf : (mat == 1) ? Wg : (mat == 2) ? Wh : Wv;
    WT[idx] = (_Float16)W[k * CH + n];
}

// ---------------------------------------------------------------------------
// Kernel 1: f/g projections (row-major f16) and v projection stored
// transposed vT[c][pixel].  One block = one 16-row tile, 8 waves = 8 col tiles.
// ---------------------------------------------------------------------------
__global__ __launch_bounds__(256) void proj_fgv_kernel(
    const float* __restrict__ x, const _Float16* __restrict__ WT,
    const float* __restrict__ bf, const float* __restrict__ bg,
    const float* __restrict__ bh,
    _Float16* __restrict__ f, _Float16* __restrict__ g,
    _Float16* __restrict__ vT) {
    const int lane = threadIdx.x & 31;
    const int wave = threadIdx.x >> 5;
    const int row0 = blockIdx.x * 16;     // global pixel-row (0..16383)
    const int n0   = wave * 16;

    const _Float16* WTf = WT;
    const _Float16* WTg = WT + 1 * CH * CH;
    const _Float16* WTh = WT + 2 * CH * CH;

    v8f cf = {}, cg = {}, cv = {};
#pragma unroll
    for (int k0 = 0; k0 < CH; k0 += 32) {
        v16h a  = load_A_f32(x, CH, row0, k0, lane);
        cf = wmma16(a, load_B_T(WTf, CH, k0, n0, lane), cf);
        cg = wmma16(a, load_B_T(WTg, CH, k0, n0, lane), cg);
        cv = wmma16(a, load_B_T(WTh, CH, k0, n0, lane), cv);
    }

    const int col = n0 + (lane & 15);     // output channel
    const int mb  = (lane >> 4) << 3;     // local row base (0 or 8)
    const float biasf = bf[col], biasg = bg[col], biash = bh[col];

    // f, g row-major: 8 stores per lane, lanes 0-15 cover 32B-contiguous runs.
#pragma unroll
    for (int r = 0; r < 8; ++r) {
        const size_t off = (size_t)(row0 + mb + r) * CH + col;
        f[off] = (_Float16)(cf[r] + biasf);
        g[off] = (_Float16)(cg[r] + biasg);
    }
    // vT transposed: each lane's 8 accumulators are consecutive M at fixed N
    // -> single contiguous 16-byte store per lane.
    {
        const int b   = row0 >> 12;                // batch
        const int mloc = (row0 & (NPIX - 1)) + mb; // pixel within batch
        _Float16* p = vT + ((size_t)b * CH + col) * NPIX + mloc;
#pragma unroll
        for (int r = 0; r < 8; ++r) p[r] = (_Float16)(cv[r] + biash);
    }
}

// ---------------------------------------------------------------------------
// Kernel 2: fused attention + output projection.
// LDS: S[16][4096] f16 (128 KB) + ctx[16][128] f32 (8 KB) + rowsum[16].
// ---------------------------------------------------------------------------
__global__ __launch_bounds__(256) void attn_fused_kernel(
    const float* __restrict__ x, const _Float16* __restrict__ WT,
    const float* __restrict__ bv, const float* __restrict__ gamma_p,
    const _Float16* __restrict__ f, const _Float16* __restrict__ g,
    const _Float16* __restrict__ vT, float* __restrict__ out) {
    extern __shared__ char smem[];
    _Float16* Sm     = (_Float16*)smem;                 // 16*4096 f16
    float*    ctx    = (float*)(smem + 16 * NPIX * 2);  // 16*128  f32
    float*    rowsum = ctx + 16 * CH;                   // 16      f32

    const int lane = threadIdx.x & 31;
    const int wave = threadIdx.x >> 5;
    const int b    = blockIdx.x >> 8;            // 256 row tiles per batch
    const int m0   = (blockIdx.x & 255) * 16;

    const _Float16* fb  = f  + (size_t)b * NPIX * CH;
    const _Float16* gb  = g  + (size_t)b * NPIX * CH;
    const _Float16* vTb = vT + (size_t)b * CH * NPIX;
    const _Float16* WTv = WT + 3 * CH * CH;

    // zero shared ctx accumulator (first use after two barriers)
    for (int i = threadIdx.x; i < 16 * CH; i += 256) ctx[i] = 0.0f;

    // query-row A fragments, reused for all 256 key tiles
    v16h af[4];
#pragma unroll
    for (int kk = 0; kk < 4; ++kk) af[kk] = load_A_f16(fb, CH, m0, kk * 32, lane);

    // ---- phase 1: S panel = f g^T; B[k][n] = g[n][k] contiguous ----------
    for (int jt = wave; jt < NPIX / 16; jt += 8) {
        const int n0 = jt * 16;
        v8f acc = {};
#pragma unroll
        for (int kk = 0; kk < 4; ++kk)
            acc = wmma16(af[kk], load_B_T(gb, CH, kk * 32, n0, lane), acc);
        const int colS = n0 + (lane & 15);
        const int mb   = (lane >> 4) << 3;
#pragma unroll
        for (int r = 0; r < 8; ++r) Sm[(mb + r) * NPIX + colS] = (_Float16)acc[r];
    }
    __syncthreads();

    // ---- phase 2: in-place softmax numerator (vectorized 8 f16 / access) -
#pragma unroll
    for (int rr = 0; rr < 2; ++rr) {
        const int m = wave * 2 + rr;
        _Float16* row = Sm + m * NPIX;
        float mx = -1e30f;
        for (int i = lane; i < NPIX / 8; i += 32) {
            const _Float16* p = row + i * 8;   // 16B aligned
#pragma unroll
            for (int e = 0; e < 8; ++e) mx = fmaxf(mx, (float)p[e]);
        }
#pragma unroll
        for (int o = 16; o >= 1; o >>= 1) mx = fmaxf(mx, __shfl_xor(mx, o, 32));
        float sum = 0.0f;
        for (int i = lane; i < NPIX / 8; i += 32) {
            _Float16* p = row + i * 8;
#pragma unroll
            for (int e = 0; e < 8; ++e) {
                const float ev = __expf((float)p[e] - mx);
                p[e] = (_Float16)ev;
                sum += ev;
            }
        }
#pragma unroll
        for (int o = 16; o >= 1; o >>= 1) sum += __shfl_xor(sum, o, 32);
        if (lane == 0) rowsum[m] = sum;
    }
    __syncthreads();

    // ---- phase 3: ctx = P @ v; B[k][n] = vT[n][k] contiguous -------------
    v8f cacc[8];
#pragma unroll
    for (int t = 0; t < 8; ++t) cacc[t] = (v8f){};
    for (int chk = wave; chk < NPIX / 32; chk += 8) {
        const int j0 = chk * 32;
        v16h aP = load_A_f16(Sm, NPIX, 0, j0, lane);
#pragma unroll
        for (int t = 0; t < 8; ++t)
            cacc[t] = wmma16(aP, load_B_T(vTb, NPIX, j0, t * 16, lane), cacc[t]);
    }
    {
        const int coln = lane & 15;
        const int mb   = (lane >> 4) << 3;
#pragma unroll
        for (int t = 0; t < 8; ++t)
#pragma unroll
            for (int r = 0; r < 8; ++r)
                atomicAdd(&ctx[(mb + r) * CH + t * 16 + coln], cacc[t][r]);
    }
    __syncthreads();

    // ---- phase 4: o = (gamma/rowsum)*ctx @ Wv + bv + x -------------------
    const float gam = gamma_p[0];
    const int n0 = wave * 16;
    v8f o = {};
#pragma unroll
    for (int kk = 0; kk < 4; ++kk) {
        const int r = lane & 15, half = lane >> 4;
        const float scale = gam / rowsum[r];
        const float* cp = ctx + r * CH;
        const int klo = kk * 32 + half * 8;
        const int khi = kk * 32 + 16 + half * 8;
        v16h a;
#pragma unroll
        for (int e = 0; e < 8; ++e) {
            a[e]     = (_Float16)(cp[klo + e] * scale);
            a[8 + e] = (_Float16)(cp[khi + e] * scale);
        }
        o = wmma16(a, load_B_T(WTv, CH, kk * 32, n0, lane), o);
    }
    const int coln = n0 + (lane & 15);
    const float bias = bv[coln];
    const int mb = (lane >> 4) << 3;
    const float* xb = x + (size_t)b * NPIX * CH;
    float* ob = out + (size_t)b * NPIX * CH;
#pragma unroll
    for (int r = 0; r < 8; ++r) {
        const size_t off = (size_t)(m0 + mb + r) * CH + coln;
        ob[off] = o[r] + bias + xb[off];
    }
}

// ---------------------------------------------------------------------------
extern "C" void kernel_launch(void* const* d_in, const int* in_sizes, int n_in,
                              void* d_out, int out_size, void* d_ws, size_t ws_size,
                              hipStream_t stream) {
    const float* x  = (const float*)d_in[0];
    const float* Wf = (const float*)d_in[1];
    const float* bf = (const float*)d_in[2];
    const float* Wg = (const float*)d_in[3];
    const float* bg = (const float*)d_in[4];
    const float* Wh = (const float*)d_in[5];
    const float* bh = (const float*)d_in[6];
    const float* Wv = (const float*)d_in[7];
    const float* bv = (const float*)d_in[8];
    const float* gm = (const float*)d_in[9];

    const size_t nElem = (size_t)BATCH * NPIX * CH;   // 2 Mi elements
    _Float16* f  = (_Float16*)d_ws;
    _Float16* g  = f + nElem;
    _Float16* vT = g + nElem;
    _Float16* WT = vT + nElem;                        // 4*128*128 f16

    // Kernel 0: one-time weight transpose + f16 convert (4*128*128 elems)
    transpose_w_kernel<<<(4 * CH * CH) / 256, 256, 0, stream>>>(Wf, Wg, Wh, Wv, WT);

    // Kernel 1: 1024 row tiles x 8 col-tile waves
    proj_fgv_kernel<<<(BATCH * NPIX) / 16, 256, 0, stream>>>(
        x, WT, bf, bg, bh, f, g, vT);

    // Kernel 2: one block per (batch, 16-row tile); ~136 KB dynamic LDS
    const size_t ldsBytes = (size_t)16 * NPIX * 2 + 16 * CH * 4 + 16 * 4;
    (void)hipFuncSetAttribute((const void*)attn_fused_kernel,
                              hipFuncAttributeMaxDynamicSharedMemorySize,
                              (int)ldsBytes);
    attn_fused_kernel<<<BATCH * (NPIX / 16), 256, ldsBytes, stream>>>(
        x, WT, bv, gm, f, g, vT, (float*)d_out);
}
